// CRFModel_15092515078861
// MI455X (gfx1250) — matrix operand order repaired and checked by reading, not compiled, not run on hardware
//
#include <hip/hip_runtime.h>
#include <hip/hip_bf16.h>
#include <stddef.h>

// CRF forward (logsumexp) via exp-space WMMA GEMM + VALU Viterbi.
// B=128, T=256, K=600 (padded to Kp=608 = 38*16 = 19*32).

#define BB   128
#define TT   256
#define KK   600
#define KP   608
#define KTIL 38    // Kp / 16 column tiles (tile index KTIL is an all-zero pad tile)
#define KCH  19    // Kp / 32 k-chunks

typedef __attribute__((ext_vector_type(16))) _Float16 v16h;
typedef __attribute__((ext_vector_type(8)))  float    v8f;

// Global (address_space(1)) pointer to v16h: forces global_load_b128 (not flat).
typedef const __attribute__((address_space(1))) v16h* gv16h_ptr;

// ---------------------------------------------------------------------------
// Prep: ETsw[(ct*KCH+kc)*32 + lane][16 halves] = exp(trans[i][j]) in a
// WMMA B-fragment (32x16 f16) layout, contiguous k like A: lane = hs*16 + n, half h <-> K = hs*16+h.
// Padding (i>=600 || j>=600) = 0.0 (exp-space -inf) so it never contributes.
// Tile ct == KTIL (j >= 608) is entirely zero: used as a clamp target so the
// tail WMMA can always execute with full EXEC.
// ---------------------------------------------------------------------------
__global__ void crf_prep_et(const float* __restrict__ trans,
                            _Float16* __restrict__ etsw) {
    const int total = (KTIL + 1) * KCH * 32 * 16;
    int idx = blockIdx.x * blockDim.x + threadIdx.x;
    if (idx >= total) return;
    int h    = idx & 15;
    int lane = (idx >> 4) & 31;
    int rem  = idx >> 9;            // ct*KCH + kc
    int kc   = rem % KCH;
    int ct   = rem / KCH;
    int i = kc * 32 + (lane >> 4) * 16 + h;   // K dim
    int j = ct * 16 + (lane & 15);            // N dim
    float v = 0.0f;
    if (i < KK && j < KK) v = __expf(trans[(size_t)i * KK + j]);
    etsw[idx] = (_Float16)v;
}

// ---------------------------------------------------------------------------
// Forward algorithm. One workgroup per 16-batch strip (8 strips), 16 waves.
// Per step: m = rowmax(alpha); P = exp(alpha - m) [f16, zero-padded];
// S = P @ expT via v_wmma_f32_16x16x32_f16; alpha' = m + log(S) + emit (masked).
// Final: logZ[b] = logsumexp_j(alpha[b, :]).
// ---------------------------------------------------------------------------
__global__ __launch_bounds__(512) void crf_forward_wmma(
    const float* __restrict__ pot, const int* __restrict__ seq_len,
    const _Float16* __restrict__ etsw, float* __restrict__ logZ) {

    __shared__ float    alphaSh[16][KP];   // 38912 B, log-space state
    __shared__ _Float16 Psh[16][640];      // 20480 B, exp-normalized (padded row)
    __shared__ float    mSh[16];
    __shared__ int      slenSh[16];

    const int strip = blockIdx.x;          // 0..7 -> batches strip*16 .. +15
    const int tid   = threadIdx.x;
    const int lane  = tid & 31;
    const int wave  = tid >> 5;            // 0..15
    const int hs    = lane >> 4;           // lane half-select
    const int ln    = lane & 15;

    if (tid < 16) slenSh[tid] = seq_len[strip * 16 + tid];

    // alpha0 = potentials[:, 0, :]; pad cols = -inf (never touched again)
    for (int e = tid; e < 16 * KP; e += 512) {
        int r = e / KP, j = e - r * KP;
        float v = -3.0e38f;
        if (j < KK) v = pot[((size_t)(strip * 16 + r) * TT) * KK + j];
        alphaSh[r][j] = v;
    }
    __syncthreads();

    // Column tiles owned by this wave; ct2 clamps into the all-zero pad tile so
    // the third WMMA is unconditional (EXEC stays all-ones, no saveexec).
    const int ct0 = wave, ct1 = wave + 16, ct2 = wave + 32;
    const int ct2c = (ct2 < KTIL) ? ct2 : KTIL;

    unsigned long long etpu = (unsigned long long)(const void*)etsw;

    for (int t = 1; t < TT; ++t) {
        // --- row max: wave w reduces row w (shuffle tree, wave32)
        {
            float mx = -3.0e38f;
            for (int j = lane; j < KK; j += 32) mx = fmaxf(mx, alphaSh[wave][j]);
            for (int off = 16; off > 0; off >>= 1)
                mx = fmaxf(mx, __shfl_xor(mx, off, 32));
            if (lane == 0) mSh[wave] = mx;
        }
        __syncthreads();

        // --- P = exp(alpha - m) as f16; zero in pad cols [600, 608)
        for (int e = tid; e < 16 * KP; e += 512) {
            int r = e / KP, j = e - r * KP;
            float p = 0.0f;
            if (j < KK) p = __expf(alphaSh[r][j] - mSh[r]);
            Psh[r][j] = (_Float16)p;
        }
        __syncthreads();

        // Opaque redefinition of the ET base pointer: stops LLVM from hoisting
        // the (t-invariant) B-fragment loads out of the t-loop (round-1 spill
        // cause). Cast back through an address_space(1) pointer so the loads
        // lower to global_load_b128 (saddr + imm), not flat_load.
        asm volatile("" : "+s"(etpu));
        gv16h_ptr etb = (gv16h_ptr)etpu;

        // --- S = P @ expT: each wave owns col tiles {w, w+16, w+32(pad)}.
        // Full unroll: all B addresses fold to saddr + lane*32 + imm, all A
        // addresses to one LDS base + imm -> near-zero address arithmetic.
        v8f acc0 = {}, acc1 = {}, acc2 = {};
#pragma unroll
        for (int kc = 0; kc < KCH; ++kc) {
            // A fragment (16x32 f16): lane ln holds row M=ln; halves 0..15 are
            // K = kc*32 + hs*16 + h.
            v16h af;
            {
                const _Float16* rp = &Psh[ln][kc * 32 + hs * 16];
#pragma unroll
                for (int h = 0; h < 8; ++h) { af[h] = rp[h]; af[8 + h] = rp[8 + h]; }
            }
            v16h b0 = etb[(ct0 * KCH + kc) * 32 + lane];
            v16h b1 = etb[(ct1 * KCH + kc) * 32 + lane];
            v16h b2 = etb[(ct2c * KCH + kc) * 32 + lane];
            acc0 = __builtin_amdgcn_wmma_f32_16x16x32_f16(
                       false, af, false, b0, (short)0, acc0, false, false);
            acc1 = __builtin_amdgcn_wmma_f32_16x16x32_f16(
                       false, af, false, b1, (short)0, acc1, false, false);
            acc2 = __builtin_amdgcn_wmma_f32_16x16x32_f16(
                       false, af, false, b2, (short)0, acc2, false, false);
        }

        // --- writeback: D layout (16x16 f32): VGPR v, lanes<16 -> M=v, else M=v+8
        {
#pragma unroll 1
            for (int ti = 0; ti < 3; ++ti) {
                int ct = (ti == 0) ? ct0 : (ti == 1) ? ct1 : ct2;
                if (ct >= KTIL) break;                 // pad tile: discard
                const v8f a = (ti == 0) ? acc0 : (ti == 1) ? acc1 : acc2;
#pragma unroll
                for (int v = 0; v < 8; ++v) {
                    int r = v + hs * 8;
                    int j = ct * 16 + ln;
                    if (j < KK) {
                        float res = mSh[r] + __logf(a[v]) +
                                    pot[((size_t)(strip * 16 + r) * TT + t) * KK + j];
                        if (t < slenSh[r]) alphaSh[r][j] = res;  // masked update
                    }
                }
            }
        }
        __syncthreads();
    }

    // --- logZ[b] = logsumexp_j alpha_final (cheap, once)
    if (tid < 16) {
        int r = tid, b = strip * 16 + r;
        float mx = -3.0e38f;
        for (int j = 0; j < KK; ++j) mx = fmaxf(mx, alphaSh[r][j]);
        float s = 0.0f;
        for (int j = 0; j < KK; ++j) s += __expf(alphaSh[r][j] - mx);
        logZ[b] = mx + __logf(s);
    }
}

// ---------------------------------------------------------------------------
// Viterbi forward: (max,+) semiring -> pure VALU. One workgroup per batch
// (128 WGs), lane j owns next-tag j; transitions read coalesced (L2-resident).
// Backpointers -> u16 workspace; identity bp on masked steps (matches ref).
// ---------------------------------------------------------------------------
__global__ __launch_bounds__(608) void crf_viterbi(
    const float* __restrict__ pot, const float* __restrict__ trans,
    const int* __restrict__ seq_len, unsigned short* __restrict__ bp,
    int* __restrict__ lastIdx) {

    __shared__ float bufA[KP];
    __shared__ float bufB[KP];
    const int b = blockIdx.x;
    const int j = threadIdx.x;            // 0..607, active j < 600
    const int slen = seq_len[b];

    float* cur = bufA;
    float* nxt = bufB;
    if (j < KK) cur[j] = pot[((size_t)b * TT) * KK + j];
    __syncthreads();

    for (int t = 1; t < TT; ++t) {
        if (j < KK) {
            float best = -3.0e38f;
            int   bi   = 0;
            const float* tc = trans + j;
#pragma unroll 4
            for (int i = 0; i < KK; ++i) {
                float v = cur[i] + tc[(size_t)i * KK];   // LDS broadcast + coalesced
                if (v > best) { best = v; bi = i; }      // strict > : first argmax
            }
            bool maskt = (t < slen);
            nxt[j] = maskt ? (best + pot[((size_t)b * TT + t) * KK + j]) : cur[j];
            bp[(size_t)(t - 1) * (BB * KK) + (size_t)b * KK + j] =
                (unsigned short)(maskt ? bi : j);
        }
        __syncthreads();
        float* tmp = cur; cur = nxt; nxt = tmp;
    }

    if (j == 0) {
        float best = cur[0];
        int   bi   = 0;
        for (int i = 1; i < KK; ++i)
            if (cur[i] > best) { best = cur[i]; bi = i; }
        lastIdx[b] = bi;
    }
}

// ---------------------------------------------------------------------------
// Backtrace + masked decode output (float, d_out[1..]).
// ---------------------------------------------------------------------------
__global__ __launch_bounds__(128) void crf_backtrace(
    const unsigned short* __restrict__ bp, const int* __restrict__ lastIdx,
    const int* __restrict__ seq_len, float* __restrict__ out) {
    int b = threadIdx.x;
    if (b >= BB) return;
    const int slen = seq_len[b];
    int cur = lastIdx[b];
    out[1 + (size_t)b * TT + (TT - 1)] = ((TT - 1) < slen) ? (float)cur : 0.0f;
    for (int t = TT - 2; t >= 0; --t) {
        cur = bp[(size_t)t * (BB * KK) + (size_t)b * KK + cur];
        out[1 + (size_t)b * TT + t] = (t < slen) ? (float)cur : 0.0f;
    }
}

// ---------------------------------------------------------------------------
// Gold-path score + loss = -mean(score - logZ) -> d_out[0].
// ---------------------------------------------------------------------------
__global__ __launch_bounds__(128) void crf_score_loss(
    const float* __restrict__ pot, const float* __restrict__ trans,
    const int* __restrict__ tags, const int* __restrict__ seq_len,
    const float* __restrict__ logZ, float* __restrict__ out) {
    __shared__ float red[BB];
    int b = threadIdx.x;
    if (b < BB) {
        const int slen = seq_len[b];
        int   prev  = tags[b * TT];
        float score = (0 < slen) ? pot[((size_t)b * TT) * KK + prev] : 0.0f;
        for (int t = 1; t < TT; ++t) {
            int tg = tags[b * TT + t];
            if (t < slen) {
                score += pot[((size_t)b * TT + t) * KK + tg];
                score += trans[(size_t)prev * KK + tg];
            }
            prev = tg;
        }
        red[b] = score - logZ[b];
    }
    __syncthreads();
    if (b == 0) {
        float s = 0.0f;
        for (int i = 0; i < BB; ++i) s += red[i];
        out[0] = -s / (float)BB;
    }
}

// ---------------------------------------------------------------------------
// Workspace layout (bytes):
//   [0)         ETsw   : 39*19*32*16 halves = 758,784  (incl. zero pad tile)
//   [758784)    logZ   : 128 f32            = 512
//   [759296)    last   : 128 i32            = 512
//   [759808)    bp     : 255*128*600 u16    = 39,168,000   (total ~39.9 MB)
// ---------------------------------------------------------------------------
extern "C" void kernel_launch(void* const* d_in, const int* in_sizes, int n_in,
                              void* d_out, int out_size, void* d_ws, size_t ws_size,
                              hipStream_t stream) {
    (void)in_sizes; (void)n_in; (void)out_size; (void)ws_size;

    const float* pot     = (const float*)d_in[0];   // (B,T,K) f32
    const float* trans   = (const float*)d_in[1];   // (K,K)   f32
    const int*   tags    = (const int*)  d_in[2];   // (B,T)   i32
    const int*   seq_len = (const int*)  d_in[3];   // (B,)    i32
    float*       out     = (float*)d_out;

    char* ws = (char*)d_ws;
    _Float16*       etsw    = (_Float16*)ws;
    float*          logZ    = (float*)(ws + 758784);
    int*            lastIdx = (int*)(ws + 759296);
    unsigned short* bp      = (unsigned short*)(ws + 759808);

    {   // pre-swizzle exp(transitions) into WMMA B-fragment layout (+pad tile)
        const int total = (KTIL + 1) * KCH * 32 * 16;
        crf_prep_et<<<(total + 255) / 256, 256, 0, stream>>>(trans, etsw);
    }
    // forward (logsumexp) recursion: 8 batch strips x 16 waves, WMMA f16
    crf_forward_wmma<<<8, 512, 0, stream>>>(pot, seq_len, etsw, logZ);
    // viterbi (max,+) recursion: 128 batches x 608 lanes
    crf_viterbi<<<BB, KP, 0, stream>>>(pot, trans, seq_len, bp, lastIdx);
    // backtrace + masked decode
    crf_backtrace<<<1, 128, 0, stream>>>(bp, lastIdx, seq_len, out);
    // sequence score + loss
    crf_score_loss<<<1, 128, 0, stream>>>(pot, trans, tags, seq_len, logZ, out);
}
